// cluster2gesture_model_1451698946508
// MI455X (gfx1250) — matrix operand order repaired and checked by reading, not compiled, not run on hardware
//
#include <hip/hip_runtime.h>

// ---------------------------------------------------------------------------
// cluster2gesture model on MI455X (gfx1250): all GEMMs via v_wmma_f32_16x16x32_bf16
// 32-row M-blocks per wave, register double-buffered k-loop.
// ---------------------------------------------------------------------------

typedef __attribute__((ext_vector_type(16))) __bf16 v16bf;
typedef __attribute__((ext_vector_type(8)))  __bf16 v8bf;
typedef __attribute__((ext_vector_type(8)))  float  v8f;

#define NB   4096   // batch
#define NT   64     // timesteps
#define NH   512    // hidden
#define ND   256    // d_out
#define NH3  1536   // 3*hidden

static __device__ __forceinline__ float sigf(float x) { return 1.0f / (1.0f + __expf(-x)); }

// A fragment (16x32 bf16, MxK): lane l -> row m0+(l&15); hi=l>>4.
// v16bf elements 0..7  = A[m][k0 + 16*hi + e]
// v16bf elements 8..15 = A[m][k0 + 16*hi + 8 + e]
static __device__ __forceinline__ v16bf a_frag(const __bf16* __restrict__ arow, int k0, int hi) {
  v8bf lo  = *(const v8bf*)(arow + k0 + hi * 16);
  v8bf hi8 = *(const v8bf*)(arow + k0 + 8 + hi * 16);
  return __builtin_shufflevector(lo, hi8, 0,1,2,3,4,5,6,7,8,9,10,11,12,13,14,15);
}

// B fragment (32x16 bf16, KxN) with B = W^T, W row-major (N x K):
// lane l -> col n0+(l&15); element e = W[n][k0 + 16*hi + e]  (16 contiguous bf16)
static __device__ __forceinline__ v16bf b_frag(const __bf16* __restrict__ wrow, int k0, int hi) {
  return *(const v16bf*)(wrow + k0 + hi * 16);
}

#define WMMA_BF16(a, b, c) \
  __builtin_amdgcn_wmma_f32_16x16x32_bf16(false, (a), false, (b), (short)0, (c), false, false)

// ---------------------------------------------------------------------------
// small helper kernels
// ---------------------------------------------------------------------------
__global__ __launch_bounds__(256) void k_convert_bf16(const float* __restrict__ src,
                                                      __bf16* __restrict__ dst, int n) {
  int i = blockIdx.x * 256 + threadIdx.x;
  if (i < n) dst[i] = (__bf16)src[i];
}

__global__ __launch_bounds__(256) void k_embed(const int* __restrict__ idx,
                                               const float* __restrict__ emb,
                                               __bf16* __restrict__ xe) {
  int i = blockIdx.x * 256 + threadIdx.x;      // i < NB*NH
  int b = i >> 9, k = i & (NH - 1);
  xe[i] = (__bf16)emb[(size_t)idx[b] * NH + k];
}

__global__ __launch_bounds__(256) void k_zero_bf16(__bf16* __restrict__ p, int n) {
  int i = blockIdx.x * 256 + threadIdx.x;
  if (i < n) p[i] = (__bf16)0.0f;
}

// outputs[:, 0, :] = 0
__global__ __launch_bounds__(256) void k_zero_out_t0(float* __restrict__ out) {
  int i = blockIdx.x * 256 + threadIdx.x;      // i < NB*ND
  int b = i >> 8, o = i & (ND - 1);
  out[(size_t)b * NT * ND + o] = 0.0f;
}

// ---------------------------------------------------------------------------
// 3-gate GEMM core: 32(M) x 16(N) x K per wave against 3 gate weight blocks.
// Double-buffered fragment loads so WMMAs overlap the next k-step's loads.
// ---------------------------------------------------------------------------
#define GATE_GEMM_CORE(A_PTR, LDA, W_PTR, K)                                    \
  const int lane = threadIdx.x & 31, wave = threadIdx.x >> 5;                   \
  const int tile = blockIdx.x * 8 + wave; /* 128 mblocks * 32 ntiles */         \
  const int mt = tile >> 5, nt = tile & 31;                                     \
  const int m0 = mt * 32, n0 = nt * 16;                                         \
  const int hi = lane >> 4, ln = lane & 15;                                     \
  const int n = n0 + ln;                                                        \
  const __bf16* arow[2] = {(A_PTR) + (size_t)(m0 + ln) * (LDA),                 \
                           (A_PTR) + (size_t)(m0 + 16 + ln) * (LDA)};           \
  const __bf16* wrow[3] = {(W_PTR) + (size_t)(0 * NH + n) * (K),                \
                           (W_PTR) + (size_t)(1 * NH + n) * (K),                \
                           (W_PTR) + (size_t)(2 * NH + n) * (K)};               \
  v8f cr[2] = {{}, {}}, cz[2] = {{}, {}}, cn[2] = {{}, {}};                     \
  v16bf a[2], b[3];                                                             \
  a[0] = a_frag(arow[0], 0, hi);                                                \
  a[1] = a_frag(arow[1], 0, hi);                                                \
  b[0] = b_frag(wrow[0], 0, hi);                                                \
  b[1] = b_frag(wrow[1], 0, hi);                                                \
  b[2] = b_frag(wrow[2], 0, hi);                                                \
  _Pragma("unroll")                                                             \
  for (int k0 = 0; k0 < (K); k0 += 32) {                                        \
    v16bf an[2], bn_[3];                                                        \
    if (k0 + 32 < (K)) {                                                        \
      an[0]  = a_frag(arow[0], k0 + 32, hi);                                    \
      an[1]  = a_frag(arow[1], k0 + 32, hi);                                    \
      bn_[0] = b_frag(wrow[0], k0 + 32, hi);                                    \
      bn_[1] = b_frag(wrow[1], k0 + 32, hi);                                    \
      bn_[2] = b_frag(wrow[2], k0 + 32, hi);                                    \
    }                                                                           \
    _Pragma("unroll")                                                           \
    for (int s = 0; s < 2; ++s) {                                               \
      cr[s] = WMMA_BF16(a[s], b[0], cr[s]);                                     \
      cz[s] = WMMA_BF16(a[s], b[1], cz[s]);                                     \
      cn[s] = WMMA_BF16(a[s], b[2], cn[s]);                                     \
    }                                                                           \
    if (k0 + 32 < (K)) {                                                        \
      a[0] = an[0]; a[1] = an[1];                                               \
      b[0] = bn_[0]; b[1] = bn_[1]; b[2] = bn_[2];                              \
    }                                                                           \
  }

// ---------------------------------------------------------------------------
// encoder: gi = xe @ pre_W_ih^T ; gates with gh = b_hh (h0 == 0); enc_h = (1-z)*n
// ---------------------------------------------------------------------------
__global__ __launch_bounds__(256)
void k_enc_gates(const __bf16* __restrict__ xe, const __bf16* __restrict__ Wih,
                 const float* __restrict__ b_ih, const float* __restrict__ b_hh,
                 float* __restrict__ ench_f, __bf16* __restrict__ ench_b) {
  GATE_GEMM_CORE(xe, NH, Wih, NH)

  const float bir = b_ih[n], biz = b_ih[NH + n], bin_ = b_ih[2 * NH + n];
  const float hr = b_hh[n], hz = b_hh[NH + n], hn = b_hh[2 * NH + n];
#pragma unroll
  for (int s = 0; s < 2; ++s)
#pragma unroll
    for (int v = 0; v < 8; ++v) {
      int mm = m0 + s * 16 + v + 8 * hi;
      float r  = sigf(cr[s][v] + bir + hr);
      float z  = sigf(cz[s][v] + biz + hz);
      float nn = tanhf(cn[s][v] + bin_ + r * hn);
      float h  = (1.0f - z) * nn;                    // + z*0
      ench_f[(size_t)mm * NH + n] = h;
      ench_b[(size_t)mm * NH + n] = (__bf16)h;
    }
}

// ---------------------------------------------------------------------------
// decoder gates: gi = x @ dec_W_ih^T + dec_b_ih ; h = (1-z)*n + z*enc_h -> bf16
// ---------------------------------------------------------------------------
__global__ __launch_bounds__(256)
void k_dec_gates(const __bf16* __restrict__ x, const __bf16* __restrict__ Wih,
                 const float* __restrict__ b_ih, const float* __restrict__ gh,
                 const float* __restrict__ ench_f, __bf16* __restrict__ h_out) {
  GATE_GEMM_CORE(x, NH, Wih, NH)

  const float bir = b_ih[n], biz = b_ih[NH + n], bin_ = b_ih[2 * NH + n];
#pragma unroll
  for (int s = 0; s < 2; ++s)
#pragma unroll
    for (int v = 0; v < 8; ++v) {
      int mm = m0 + s * 16 + v + 8 * hi;
      const float* ghrow = gh + (size_t)mm * NH3;
      float r  = sigf(cr[s][v] + bir + ghrow[n]);
      float z  = sigf(cz[s][v] + biz + ghrow[NH + n]);
      float nn = tanhf(cn[s][v] + bin_ + r * ghrow[2 * NH + n]);
      float h  = (1.0f - z) * nn + z * ench_f[(size_t)mm * NH + n];
      h_out[(size_t)mm * NH + n] = (__bf16)h;
    }
}

// ---------------------------------------------------------------------------
// plain GEMM core: 32(M) x 32(N) x K per wave, double-buffered.
// ---------------------------------------------------------------------------
#define PLAIN_GEMM_CORE(A_PTR, LDA, W_PTR, K, NTILES)                           \
  const int lane = threadIdx.x & 31, wave = threadIdx.x >> 5;                   \
  const int tile = blockIdx.x * 8 + wave;                                       \
  const int mt = tile / (NTILES), nt = tile % (NTILES);                         \
  const int m0 = mt * 32, n0 = nt * 32;                                         \
  const int hi = lane >> 4, ln = lane & 15;                                     \
  const __bf16* arow[2] = {(A_PTR) + (size_t)(m0 + ln) * (LDA),                 \
                           (A_PTR) + (size_t)(m0 + 16 + ln) * (LDA)};           \
  const __bf16* wrow[2] = {(W_PTR) + (size_t)(n0 + ln) * (K),                   \
                           (W_PTR) + (size_t)(n0 + 16 + ln) * (K)};             \
  v8f c[2][2] = {{{}, {}}, {{}, {}}};                                           \
  v16bf a[2], b[2];                                                             \
  a[0] = a_frag(arow[0], 0, hi);                                                \
  a[1] = a_frag(arow[1], 0, hi);                                                \
  b[0] = b_frag(wrow[0], 0, hi);                                                \
  b[1] = b_frag(wrow[1], 0, hi);                                                \
  _Pragma("unroll")                                                             \
  for (int k0 = 0; k0 < (K); k0 += 32) {                                        \
    v16bf an[2], bn_[2];                                                        \
    if (k0 + 32 < (K)) {                                                        \
      an[0]  = a_frag(arow[0], k0 + 32, hi);                                    \
      an[1]  = a_frag(arow[1], k0 + 32, hi);                                    \
      bn_[0] = b_frag(wrow[0], k0 + 32, hi);                                    \
      bn_[1] = b_frag(wrow[1], k0 + 32, hi);                                    \
    }                                                                           \
    _Pragma("unroll")                                                           \
    for (int s = 0; s < 2; ++s) {                                               \
      c[s][0] = WMMA_BF16(a[s], b[0], c[s][0]);                                 \
      c[s][1] = WMMA_BF16(a[s], b[1], c[s][1]);                                 \
    }                                                                           \
    if (k0 + 32 < (K)) {                                                        \
      a[0] = an[0]; a[1] = an[1]; b[0] = bn_[0]; b[1] = bn_[1];                 \
    }                                                                           \
  }

// ---------------------------------------------------------------------------
// gh = enc_h @ dec_W_hh^T + dec_b_hh    (4096 x 1536, K=512), fp32 out
// ---------------------------------------------------------------------------
__global__ __launch_bounds__(256)
void k_gh(const __bf16* __restrict__ ench_b, const __bf16* __restrict__ Whh,
          const float* __restrict__ b_hh, float* __restrict__ gh) {
  PLAIN_GEMM_CORE(ench_b, NH, Whh, NH, 48)   // 128 mblocks * 48 nblocks

#pragma unroll
  for (int s = 0; s < 2; ++s)
#pragma unroll
    for (int u = 0; u < 2; ++u) {
      const int nn_col = n0 + u * 16 + ln;
      const float bias = b_hh[nn_col];
#pragma unroll
      for (int v = 0; v < 8; ++v) {
        int mm = m0 + s * 16 + v + 8 * hi;
        gh[(size_t)mm * NH3 + nn_col] = c[s][u][v] + bias;
      }
    }
}

// ---------------------------------------------------------------------------
// x = relu(BN(dec_in @ lin_W^T + lin_b))   (4096 x 512, K=256) -> bf16
// ---------------------------------------------------------------------------
__global__ __launch_bounds__(256)
void k_lin_bn(const __bf16* __restrict__ dec_in, const __bf16* __restrict__ Wlin,
              const float* __restrict__ lin_b, const float* __restrict__ gamma,
              const float* __restrict__ beta, const float* __restrict__ mean,
              const float* __restrict__ var, __bf16* __restrict__ x_out) {
  PLAIN_GEMM_CORE(dec_in, ND, Wlin, ND, 16)  // 128 mblocks * 16 nblocks

#pragma unroll
  for (int u = 0; u < 2; ++u) {
    const int nn_col = n0 + u * 16 + ln;
    const float bias = lin_b[nn_col];
    const float sc = gamma[nn_col] * rsqrtf(var[nn_col] + 1e-5f);
    const float mu = mean[nn_col], bt = beta[nn_col];
#pragma unroll
    for (int s = 0; s < 2; ++s)
#pragma unroll
      for (int v = 0; v < 8; ++v) {
        int mm = m0 + s * 16 + v + 8 * hi;
        float val = (c[s][u][v] + bias - mu) * sc + bt;
        x_out[(size_t)mm * NH + nn_col] = (__bf16)fmaxf(val, 0.0f);
      }
  }
}

// ---------------------------------------------------------------------------
// out = h @ out_W^T + out_b  (4096 x 256, K=512) -> fp32 outputs[:,t,:] + bf16 dec_in
// ---------------------------------------------------------------------------
__global__ __launch_bounds__(256)
void k_out(const __bf16* __restrict__ h, const __bf16* __restrict__ Wout,
           const float* __restrict__ b_out, float* __restrict__ out, int t,
           __bf16* __restrict__ dec_next) {
  PLAIN_GEMM_CORE(h, NH, Wout, NH, 8)        // 128 mblocks * 8 nblocks

#pragma unroll
  for (int u = 0; u < 2; ++u) {
    const int nn_col = n0 + u * 16 + ln;
    const float bias = b_out[nn_col];
#pragma unroll
    for (int s = 0; s < 2; ++s)
#pragma unroll
      for (int v = 0; v < 8; ++v) {
        int mm = m0 + s * 16 + v + 8 * hi;
        float o = c[s][u][v] + bias;
        out[(size_t)mm * (NT * ND) + (size_t)t * ND + nn_col] = o;
        dec_next[(size_t)mm * ND + nn_col] = (__bf16)o;
      }
  }
}

// ---------------------------------------------------------------------------
extern "C" void kernel_launch(void* const* d_in, const int* in_sizes, int n_in,
                              void* d_out, int out_size, void* d_ws, size_t ws_size,
                              hipStream_t stream) {
  const int*   idx   = (const int*)d_in[0];
  // d_in[1] = out_poses (shape only, unused)
  const float* emb   = (const float*)d_in[2];
  const float* pWih  = (const float*)d_in[3];
  // d_in[4] = pre_W_hh (unused: h0 == 0)
  const float* pbih  = (const float*)d_in[5];
  const float* pbhh  = (const float*)d_in[6];
  const float* linW  = (const float*)d_in[7];
  const float* linb  = (const float*)d_in[8];
  const float* gam   = (const float*)d_in[9];
  const float* bet   = (const float*)d_in[10];
  const float* mu    = (const float*)d_in[11];
  const float* var   = (const float*)d_in[12];
  const float* dWih  = (const float*)d_in[13];
  const float* dWhh  = (const float*)d_in[14];
  const float* dbih  = (const float*)d_in[15];
  const float* dbhh  = (const float*)d_in[16];
  const float* outW  = (const float*)d_in[17];
  const float* outb  = (const float*)d_in[18];
  float* out = (float*)d_out;

  char* ws = (char*)d_ws;
  size_t off = 0;
  auto alloc = [&](size_t bytes) -> char* {
    char* p = ws + off;
    off += (bytes + 255) & ~(size_t)255;
    return p;
  };
  __bf16* wih_pre = (__bf16*)alloc((size_t)NH3 * NH * 2);
  __bf16* whh_dec = (__bf16*)alloc((size_t)NH3 * NH * 2);
  __bf16* wih_dec = (__bf16*)alloc((size_t)NH3 * NH * 2);
  __bf16* linw_b  = (__bf16*)alloc((size_t)NH * ND * 2);
  __bf16* outw_b  = (__bf16*)alloc((size_t)ND * NH * 2);
  __bf16* xe      = (__bf16*)alloc((size_t)NB * NH * 2);
  float*  ench_f  = (float*) alloc((size_t)NB * NH * 4);
  __bf16* ench_b  = (__bf16*)alloc((size_t)NB * NH * 2);
  float*  gh      = (float*) alloc((size_t)NB * NH3 * 4);
  __bf16* xbuf    = (__bf16*)alloc((size_t)NB * NH * 2);
  __bf16* hbuf    = (__bf16*)alloc((size_t)NB * NH * 2);
  __bf16* dec_in  = (__bf16*)alloc((size_t)NB * ND * 2);

  // --- weight conversions (fp32 -> bf16), once per launch ---
  {
    int n = NH3 * NH;
    k_convert_bf16<<<(n + 255) / 256, 256, 0, stream>>>(pWih, wih_pre, n);
    k_convert_bf16<<<(n + 255) / 256, 256, 0, stream>>>(dWhh, whh_dec, n);
    k_convert_bf16<<<(n + 255) / 256, 256, 0, stream>>>(dWih, wih_dec, n);
    n = NH * ND;
    k_convert_bf16<<<(n + 255) / 256, 256, 0, stream>>>(linW, linw_b, n);
    k_convert_bf16<<<(n + 255) / 256, 256, 0, stream>>>(outW, outw_b, n);
  }

  // --- encoder ---
  k_embed<<<(NB * NH) / 256, 256, 0, stream>>>(idx, emb, xe);
  k_enc_gates<<<(NB / 32) * (NH / 16) / 8, 256, 0, stream>>>(xe, wih_pre, pbih, pbhh,
                                                             ench_f, ench_b);
  k_gh<<<(NB / 32) * (NH3 / 32) / 8, 256, 0, stream>>>(ench_b, whh_dec, dbhh, gh);

  // --- init carried state and outputs[:,0,:] ---
  k_zero_bf16<<<(NB * ND) / 256, 256, 0, stream>>>(dec_in, NB * ND);
  k_zero_out_t0<<<(NB * ND) / 256, 256, 0, stream>>>(out);

  // --- sequential decoder: 63 steps, dep chain through dec_in ---
  for (int t = 1; t < NT; ++t) {
    k_lin_bn<<<(NB / 32) * (NH / 32) / 8, 256, 0, stream>>>(dec_in, linw_b, linb, gam,
                                                            bet, mu, var, xbuf);
    k_dec_gates<<<(NB / 32) * (NH / 16) / 8, 256, 0, stream>>>(xbuf, wih_dec, dbih, gh,
                                                               ench_f, hbuf);
    k_out<<<(NB / 32) * (ND / 32) / 8, 256, 0, stream>>>(hbuf, outw_b, outb, out, t,
                                                         dec_in);
  }
}
